// VectorQuantizer_84172769068255
// MI455X (gfx1250) — matrix operand order repaired and checked by reading, not compiled, not run on hardware
//
#include <hip/hip_runtime.h>
#include <hip/hip_bf16.h>

typedef _Float16 v16h  __attribute__((ext_vector_type(16)));
typedef _Float16 f16x8 __attribute__((ext_vector_type(8)));
typedef float    v8f   __attribute__((ext_vector_type(8)));

#define D_DIM   256
#define K_CODES 1024
#define N_ROWS  65536          // 16 * 64 * 64
#define N_ELEM  16777216       // 16 * 256 * 64 * 64
#define MT      128            // rows per block (8 waves x 16 rows)
#define KT      64             // codes per K-tile
#define NTILES  (K_CODES / KT) // 16
#define LDSP    264            // padded LDS row stride in f16 (528B -> 4-bank step)
#define TILE_BYTES (KT * D_DIM * 2)   // 32768 B of f16 codes per tile

// workspace layout (bytes)
#define WS_COUNTS 0            // int[1024]
#define WS_E2     4096         // float[1024]
#define WS_SUMSQ  8192         // double
#define WS_IDX    12288        // int[65536]
#define WS_EMBH   274432       // _Float16[1024*256] = 524288 B

// ---------------------------------------------------------------------------
// Prep: e2[k] = ||emb_k||^2, f16 codebook copy, zero histogram/accumulator
// ---------------------------------------------------------------------------
__global__ __launch_bounds__(256) void vq_prep(const float* __restrict__ emb,
                                               float* __restrict__ e2,
                                               _Float16* __restrict__ emb_h,
                                               int* __restrict__ counts,
                                               double* __restrict__ sumsq) {
    int k = blockIdx.x * 256 + threadIdx.x;          // 4 blocks -> k in [0,1024)
    const float* row = emb + k * D_DIM;
    _Float16* hrow = emb_h + k * D_DIM;
    float s = 0.f;
#pragma unroll 8
    for (int d = 0; d < D_DIM; ++d) {
        float v = row[d];
        s += v * v;
        hrow[d] = (_Float16)v;
    }
    e2[k] = s;
    counts[k] = 0;
    if (k == 0) *sumsq = 0.0;
}

// ---------------------------------------------------------------------------
// Fused GEMM (f16 WMMA, f32 accum) + per-row argmin + histogram.
// B tiles are double-buffered in LDS and filled with async global->LDS DMA
// (ASYNCcnt) so HBM/L2 traffic overlaps the WMMA stream.
// ---------------------------------------------------------------------------
__global__ __launch_bounds__(256) void vq_argmin(const float* __restrict__ x,
                                                 const _Float16* __restrict__ emb_h,
                                                 const float* __restrict__ e2,
                                                 int* __restrict__ idx,
                                                 int* __restrict__ counts) {
    __shared__ __align__(16) _Float16 As[MT * LDSP];      // 67,584 B
    __shared__ __align__(16) _Float16 Bs[2][KT * LDSP];   // 67,584 B
    __shared__ float e2s[K_CODES];                        //  4,096 B

    const int t   = threadIdx.x;
    const int n0  = blockIdx.x * MT;     // MT divides 4096, so one image per block
    const int b   = n0 >> 12;
    const int hw0 = n0 & 4095;

    // issue one B tile as 2048 x 16B async copies (8 per thread).
    // chunk j: global byte = j*16 (contiguous); LDS byte = (j>>5)*528 + (j&31)*16
    auto issue_tile = [&](int kt, int buf) {
        const unsigned lbase = (unsigned)(unsigned long long)&Bs[buf][0];
        const void* gbase = (const char*)emb_h + kt * TILE_BYTES;
#pragma unroll
        for (int it = 0; it < 8; ++it) {
            const int j = t + (it << 8);
            const unsigned lds  = lbase + (unsigned)((j >> 5) * (LDSP * 2)) + ((j & 31) << 4);
            const unsigned goff = (unsigned)(j << 4);
            asm volatile("global_load_async_to_lds_b128 %0, %1, %2"
                         :: "v"(lds), "v"(goff), "s"(gbase)
                         : "memory");
        }
    };

    issue_tile(0, 0);                    // prefetch first tile while staging A

    // stage x tile as f16 (rows = n, cols = channel); float4 along coalesced hw
    for (int j = t; j < (MT * D_DIM) / 4; j += 256) {
        int d = j >> 5;                  // channel 0..255
        int r = (j & 31) << 2;           // row group of 4
        float4 v = *(const float4*)&x[((b * D_DIM + d) << 12) + hw0 + r];
        As[(r + 0) * LDSP + d] = (_Float16)v.x;
        As[(r + 1) * LDSP + d] = (_Float16)v.y;
        As[(r + 2) * LDSP + d] = (_Float16)v.z;
        As[(r + 3) * LDSP + d] = (_Float16)v.w;
    }
    for (int i = t; i < K_CODES; i += 256) e2s[i] = e2[i];

    const int lane = t & 31;
    const int w    = t >> 5;             // wave id: owns rows 16w..16w+15
    const int h    = lane >> 4;
    const int l15  = lane & 15;
    const int rowA = (w << 4) + l15;     // A-fragment row (both lane halves)

    // running per-lane packed minima: key = mono(dist)[31:10] | col[9:0]
    unsigned int best[8];
#pragma unroll
    for (int r = 0; r < 8; ++r) best[r] = 0xFFFFFFFFu;

    const v8f vzero = {};

    for (int kt = 0; kt < NTILES; ++kt) {
        const int cur = kt & 1;
        // my async copies for tile kt are done; barrier makes everyone's visible
        // and closes the read window on the buffer refilled next.
        asm volatile("s_wait_asynccnt 0x0" ::: "memory");
        __syncthreads();
        if (kt + 1 < NTILES) issue_tile(kt + 1, cur ^ 1);

        const _Float16* bs = &Bs[cur][0];

        v8f acc[4];
#pragma unroll
        for (int ct = 0; ct < 4; ++ct) acc[ct] = vzero;

#pragma unroll
        for (int ks = 0; ks < D_DIM / 32; ++ks) {
            const int kb = ks * 32;
            // A fragment: elem j<8 -> K=kb+8h+j ; j>=8 -> K=kb+16+8h+(j-8)
            const f16x8 a0 = *(const f16x8*)&As[rowA * LDSP + kb + 8 * h];
            const f16x8 a1 = *(const f16x8*)&As[rowA * LDSP + kb + 16 + 8 * h];
            const v16h  a  = __builtin_shufflevector(a0, a1,
                                0,1,2,3,4,5,6,7,8,9,10,11,12,13,14,15);
            // batch-load all four B fragments so the scheduler can use
            // partial s_wait_dscnt instead of a full wait before each WMMA
            v16h bf[4];
#pragma unroll
            for (int ct = 0; ct < 4; ++ct) {
                const int code = (ct << 4) + l15;
                const f16x8 b0 = *(const f16x8*)&bs[code * LDSP + kb + 8 * h];
                const f16x8 b1 = *(const f16x8*)&bs[code * LDSP + kb + 16 + 8 * h];
                bf[ct] = __builtin_shufflevector(b0, b1,
                              0,1,2,3,4,5,6,7,8,9,10,11,12,13,14,15);
            }
#pragma unroll
            for (int ct = 0; ct < 4; ++ct) {
                acc[ct] = __builtin_amdgcn_wmma_f32_16x16x32_f16(
                              false, a, false, bf[ct], (short)0, acc[ct], false, false);
            }
        }

        // distance epilogue: fold candidates into per-lane packed minima.
        // No cross-lane traffic here -- each lane position owns a fixed row.
#pragma unroll
        for (int ct = 0; ct < 4; ++ct) {
            const int   col = kt * KT + (ct << 4) + l15;
            const float ev  = e2s[col];
#pragma unroll
            for (int r = 0; r < 8; ++r) {
                float dv = fmaf(-2.0f, acc[ct][r], ev);
                unsigned int bits = __float_as_uint(dv);
                // monotonic map: float total order -> unsigned total order
                unsigned int mono = bits ^ ((unsigned int)((int)bits >> 31) | 0x80000000u);
                unsigned int key  = (mono & 0xFFFFFC00u) | (unsigned int)col;
                best[r] = (key < best[r]) ? key : best[r];
            }
        }
    }

    // single final 16-lane butterfly min per row
#pragma unroll
    for (int r = 0; r < 8; ++r) {
        unsigned int k = best[r];
#pragma unroll
        for (int m = 1; m <= 8; m <<= 1) {
            unsigned int o = (unsigned int)__shfl_xor((int)k, m, 32);
            k = (o < k) ? o : k;
        }
        best[r] = k;
    }

    // C-layout: VGPR r holds row M = r + 8h; all 16 lanes of a half agree
    if (l15 == 0) {
#pragma unroll
        for (int r = 0; r < 8; ++r) {
            int row = n0 + (w << 4) + (h << 3) + r;
            int ci  = (int)(best[r] & 1023u);
            idx[row] = ci;
            atomicAdd(&counts[ci], 1);
        }
    }
}

// ---------------------------------------------------------------------------
// Gather codes back to [B,C,H,W] + accumulate squared error
// ---------------------------------------------------------------------------
__global__ __launch_bounds__(256) void vq_gather(const float* __restrict__ x,
                                                 const float* __restrict__ emb,
                                                 const int* __restrict__ idx,
                                                 float* __restrict__ out,
                                                 double* __restrict__ sumsq) {
    __shared__ float red[256];
    int i  = blockIdx.x * 256 + threadIdx.x;     // < 16,777,216
    int c  = (i >> 12) & 255;
    int bb = i >> 20;
    int hw = i & 4095;
    int n  = (bb << 12) | hw;
    int k  = idx[n];
    float q  = emb[k * D_DIM + c];
    float xv = x[i];
    out[i] = q;                                   // straight-through value == quantized
    float df = q - xv;
    red[threadIdx.x] = df * df;
    __syncthreads();
#pragma unroll
    for (int s = 128; s > 0; s >>= 1) {
        if (threadIdx.x < s) red[threadIdx.x] += red[threadIdx.x + s];
        __syncthreads();
    }
    if (threadIdx.x == 0) atomicAdd(sumsq, (double)red[0]);
}

// ---------------------------------------------------------------------------
// Finalize: loss scalar + perplexity from histogram
// ---------------------------------------------------------------------------
__global__ __launch_bounds__(1024) void vq_final(const int* __restrict__ counts,
                                                 const double* __restrict__ sumsq,
                                                 float* __restrict__ d_out,
                                                 int out_size) {
    __shared__ float red[1024];
    int k = threadIdx.x;
    float p = (float)counts[k] / (float)N_ROWS;
    red[k] = p * logf(p + 1e-10f);
    __syncthreads();
#pragma unroll
    for (int s = 512; s > 0; s >>= 1) {
        if (k < s) red[k] += red[k + s];
        __syncthreads();
    }
    if (k == 0) {
        d_out[0]            = (float)(1.25 * (*sumsq) / ((double)N_ROWS * (double)D_DIM));
        d_out[out_size - 1] = expf(-red[0]);
    }
}

// ---------------------------------------------------------------------------
extern "C" void kernel_launch(void* const* d_in, const int* in_sizes, int n_in,
                              void* d_out, int out_size, void* d_ws, size_t ws_size,
                              hipStream_t stream) {
    const float* x   = (const float*)d_in[0];    // [16,256,64,64]
    const float* emb = (const float*)d_in[1];    // [1024,256]
    float* out = (float*)d_out;                  // [loss, out(16.7M), perplexity]
    char*  ws  = (char*)d_ws;

    int*      counts = (int*)(ws + WS_COUNTS);
    float*    e2     = (float*)(ws + WS_E2);
    double*   sumsq  = (double*)(ws + WS_SUMSQ);
    int*      idx    = (int*)(ws + WS_IDX);
    _Float16* emb_h  = (_Float16*)(ws + WS_EMBH);

    vq_prep  <<<K_CODES / 256, 256, 0, stream>>>(emb, e2, emb_h, counts, sumsq);
    vq_argmin<<<N_ROWS / MT,   256, 0, stream>>>(x, emb_h, e2, idx, counts);
    vq_gather<<<N_ELEM / 256,  256, 0, stream>>>(x, emb, idx, out + 1, sumsq);
    vq_final <<<1,            1024, 0, stream>>>(counts, sumsq, out, out_size);
}
